// EdgeUpdateBlock_9131100471461
// MI455X (gfx1250) — matrix operand order repaired and checked by reading, not compiled, not run on hardware
//
#include <hip/hip_runtime.h>

typedef __attribute__((ext_vector_type(16))) __bf16        v16bf;
typedef __attribute__((ext_vector_type(8)))  float         v8f;
typedef __attribute__((ext_vector_type(8)))  unsigned int  v8u;

#define DD    64      // node / edge feature dim
#define INF   192     // concat width (2*D + D)
#define HH    128     // hidden (2*C)
#define CO    64      // output C
#define NW    4       // waves per block (wave32)
#define BLOCK (NW * 32)

// fp32 pair -> packed bf16 dword using the CDNA5 hardware packed convert.
// Pure (no volatile / no memory clobber) so the scheduler may reorder/CSE it.
__device__ __forceinline__ unsigned int pack2(float a, float b) {
  unsigned int d;
  asm("v_cvt_pk_bf16_f32 %0, %1, %2" : "=v"(d) : "v"(a), "v"(b));
  return d;
}

// shifted softplus via raw v_exp_f32 / v_log_f32 (both are base-2 on AMDGCN)
__device__ __forceinline__ float ssoftplus(float v) {
#if __has_builtin(__builtin_amdgcn_exp2f) && __has_builtin(__builtin_amdgcn_logf)
  const float LOG2E = 1.4426950408889634f;
  const float LN2   = 0.6931471805599453f;
  float e = __builtin_amdgcn_exp2f(-fabsf(v) * LOG2E);   // exp(-|v|)
  float l = __builtin_amdgcn_logf(1.0f + e);             // log2(1+e)
  return fmaf(LN2, l, fmaxf(v, 0.0f) - LN2);             // max(v,0)+ln(1+e)-ln2
#else
  return fmaxf(v, 0.0f) + __logf(1.0f + __expf(-fabsf(v))) - 0.6931471805599453f;
#endif
}

// build a 16-wide bf16 A-fragment from 16 consecutive-K fp32 values
// (elems 0-15 at p[0..15]); p must be 32B aligned
__device__ __forceinline__ v16bf make_afrag(const float* p) {
  float4 x0 = *(const float4*)(p);
  float4 x1 = *(const float4*)(p + 4);
  float4 y0 = *(const float4*)(p + 8);
  float4 y1 = *(const float4*)(p + 12);
  v8u a;
  a[0] = pack2(x0.x, x0.y); a[1] = pack2(x0.z, x0.w);
  a[2] = pack2(x1.x, x1.y); a[3] = pack2(x1.z, x1.w);
  a[4] = pack2(y0.x, y0.y); a[5] = pack2(y0.z, y0.w);
  a[6] = pack2(y1.x, y1.y); a[7] = pack2(y1.z, y1.w);
  return __builtin_bit_cast(v16bf, a);
}

__global__ __launch_bounds__(BLOCK) void edge_mlp_wmma(
    const float* __restrict__ h, const float* __restrict__ ea,
    const long long* __restrict__ eidx,          // (2, E) int64
    const float* __restrict__ W1, const float* __restrict__ b1,
    const float* __restrict__ W2, const float* __restrict__ b2,
    float* __restrict__ out, int E, int ntiles)
{
  // bf16 weights, transposed so B-fragments are K-contiguous per output column
  __shared__ __align__(64) unsigned short sW1T[HH * INF];   // [n][k] 48 KB
  __shared__ __align__(64) unsigned short sW2T[CO * HH];    // [n][k] 16 KB
  __shared__ __align__(64) float sAct[NW][16 * HH];         // fp32 16x128/wave, 32 KB

  const int t = threadIdx.x;
  for (int i = t; i < INF * HH; i += BLOCK) {       // W1 is (192,128) row-major
    int k = i / HH, n = i % HH;
    sW1T[n * INF + k] = (unsigned short)(pack2(W1[i], 0.0f) & 0xFFFFu);
  }
  for (int i = t; i < HH * CO; i += BLOCK) {        // W2 is (128,64) row-major
    int k = i / CO, n = i % CO;
    sW2T[n * HH + k] = (unsigned short)(pack2(W2[i], 0.0f) & 0xFFFFu);
  }
  __syncthreads();

  const int wave = t >> 5;
  const int lane = t & 31;
  const int lo   = lane & 15;   // column / row-within-tile index
  const int hi   = lane >> 4;   // half-wave select (ISA K/M split)

  float bias1[8], bias2[4];
#pragma unroll
  for (int nt = 0; nt < 8; ++nt) bias1[nt] = b1[nt * 16 + lo];
#pragma unroll
  for (int nt = 0; nt < 4; ++nt) bias2[nt] = b2[nt * 16 + lo];

  float* act = &sAct[wave][0];
  const int kw_lane = hi * 16;  // B-fragment K base: lanes 0-15 K=0..15, 16-31 K=16..31

  for (int tile = blockIdx.x * NW + wave; tile < ntiles; tile += gridDim.x * NW) {
    const int ebase = tile * 16;
    int e = ebase + lo;
    if (e >= E) e = E - 1;      // branchless clamp: EXEC stays all-ones for WMMA

    const long long srcn = eidx[e];
    const long long dstn = eidx[(size_t)E + e];
    const float* r0 = h  + (size_t)srcn * DD;   // concat cols   0.. 63
    const float* r1 = h  + (size_t)dstn * DD;   // concat cols  64..127
    const float* r2 = ea + (size_t)e    * DD;   // concat cols 128..191

    // speculative prefetch of the next tile's edge_attr row (global_prefetch_b8)
    __builtin_prefetch(ea + ((size_t)(tile + gridDim.x * NW) * 16 + lo) * DD, 0, 0);

    // ---------------- layer 1: (16x192) @ (192x128), bf16 WMMA, f32 acc ----
    v8f acc1[8];
#pragma unroll
    for (int nt = 0; nt < 8; ++nt)
#pragma unroll
      for (int r = 0; r < 8; ++r) acc1[nt][r] = bias1[nt];

#pragma unroll
    for (int kt = 0; kt < 6; ++kt) {
      const float* p = (kt < 2) ? r0 : (kt < 4) ? r1 : r2;
      const int c0 = (kt & 1) * 32 + hi * 16;  // A layout: elems 0-15 K=hi*16+j
      v16bf av = make_afrag(p + c0);
      const int kb = kt * 32 + kw_lane;
#pragma unroll
      for (int nt = 0; nt < 8; ++nt) {
        v16bf bv = *(const v16bf*)&sW1T[(nt * 16 + lo) * INF + kb];
        acc1[nt] = __builtin_amdgcn_wmma_f32_16x16x32_bf16(
            false, av, false, bv, (short)0, acc1[nt], false, false);
      }
    }

    // ------- shifted softplus, stage fp32 16x128 tile to LDS (N->K transpose)
#pragma unroll
    for (int nt = 0; nt < 8; ++nt) {
      const int n = nt * 16 + lo;               // C/D layout: lane owns column N
#pragma unroll
      for (int r = 0; r < 8; ++r)               // VGPR r holds row M = r + hi*8
        act[(r + hi * 8) * HH + n] = ssoftplus(acc1[nt][r]);
    }
    asm volatile("s_wait_dscnt 0" ::: "memory");  // wave-local LDS RAW before re-read

    // ---------------- layer 2: (16x128) @ (128x64) ------------------------
    v8f acc2[4];
#pragma unroll
    for (int nt = 0; nt < 4; ++nt)
#pragma unroll
      for (int r = 0; r < 8; ++r) acc2[nt][r] = bias2[nt];

#pragma unroll
    for (int kt = 0; kt < 4; ++kt) {
      // A' fragment: row m = lo, K = kt*32 + hi*16 + {0..15}, fp32 in LDS
      v16bf av = make_afrag(act + lo * HH + kt * 32 + hi * 16);
      const int kb = kt * 32 + kw_lane;
#pragma unroll
      for (int nt = 0; nt < 4; ++nt) {
        v16bf bv = *(const v16bf*)&sW2T[(nt * 16 + lo) * HH + kb];
        acc2[nt] = __builtin_amdgcn_wmma_f32_16x16x32_bf16(
            false, av, false, bv, (short)0, acc2[nt], false, false);
      }
    }

    // ---------------- store 16x64 fp32 output tile ------------------------
#pragma unroll
    for (int nt = 0; nt < 4; ++nt) {
      const int n = nt * 16 + lo;
#pragma unroll
      for (int r = 0; r < 8; ++r) {
        const int m = ebase + r + hi * 8;
        if (m < E) out[(size_t)m * CO + n] = acc2[nt][r];
      }
    }
  }
}

extern "C" void kernel_launch(void* const* d_in, const int* in_sizes, int n_in,
                              void* d_out, int out_size, void* d_ws, size_t ws_size,
                              hipStream_t stream) {
  (void)n_in; (void)out_size; (void)d_ws; (void)ws_size;
  const float*     h    = (const float*)d_in[0];
  const float*     ea   = (const float*)d_in[1];
  const long long* eidx = (const long long*)d_in[2];  // int64 (2,E)
  const float*     W1   = (const float*)d_in[3];
  const float*     b1   = (const float*)d_in[4];
  const float*     W2   = (const float*)d_in[5];
  const float*     b2   = (const float*)d_in[6];
  float*           out  = (float*)d_out;

  const int E      = in_sizes[1] / DD;     // edge_attr is (E, 64)
  const int ntiles = (E + 15) / 16;
  const int blocks = (ntiles + NW - 1) / NW;

  edge_mlp_wmma<<<blocks, BLOCK, 0, stream>>>(h, ea, eidx, W1, b1, W2, b2,
                                              out, E, ntiles);
}
